// TLSTM_59966333387415
// MI455X (gfx1250) — matrix operand order repaired and checked by reading, not compiled, not run on hardware
//
#include <hip/hip_runtime.h>
#include <hip/hip_bf16.h>
#include <math.h>

#define INPUT_SIZE 70
#define HIDDEN     2048
#define T_STEPS    4096
#define G4         (4 * HIDDEN)        // 8192 gate rows
#define KCAT       2144                // 2048 + 70, padded to 67*32
#define KT_CAT     67                  // k-tiles (32 wide) for gates
#define KT_WC      64                  // k-tiles for Wc
#define RT_CAT     (G4 / 16)           // 512 row-tiles for gates
#define RT_WC      (HIDDEN / 16)       // 128 row-tiles for Wc
#define NWG        80                  // 80 WGs * 8 waves = 640 waves = 512+128 jobs
#define TPB        256
#define WPW        8                   // waves per workgroup (wave32)
#define TILE_ELEMS 512                 // 32 lanes * 16 bf16 per B tile (1 KB)

typedef __attribute__((ext_vector_type(16))) __bf16 v16bf;
typedef __attribute__((ext_vector_type(8)))  __bf16 v8bf;
typedef __attribute__((ext_vector_type(8)))  float  v8f;

static __device__ __forceinline__ unsigned short f2bf(float f) {
  unsigned u = __float_as_uint(f);
  unsigned r = 0x7FFFu + ((u >> 16) & 1u);   // round-to-nearest-even
  return (unsigned short)((u + r) >> 16);
}

static __device__ __forceinline__ float sigmoidf_(float x) {
  return 1.0f / (1.0f + __expf(-x));
}

// Monotonic-phase grid barrier: counter only ever increases -> no reset race.
static __device__ __forceinline__ void grid_barrier(unsigned* cnt, unsigned phase) {
  __threadfence();               // make this thread's global writes agent-visible
  __syncthreads();
  if (threadIdx.x == 0) {
    __hip_atomic_fetch_add(cnt, 1u, __ATOMIC_ACQ_REL, __HIP_MEMORY_SCOPE_AGENT);
    const unsigned target = phase * NWG;
    while (__hip_atomic_load(cnt, __ATOMIC_ACQUIRE, __HIP_MEMORY_SCOPE_AGENT) < target) {
      __builtin_amdgcn_s_sleep(2);
    }
  }
  __syncthreads();
}

// ---------------------------------------------------------------------------
// Kernel 1: convert + transpose weights into WMMA B-tile layout (bf16).
// Tile (rt, kt) holds W[rt*16 + n, kt*32 + k] for n in [0,16), k in [0,32),
// stored so lane L reads its 16 bf16 (32 B) contiguously:
//   lane < 16 : column n = lane,    K = 0..15  of the tile
//   lane >= 16: column n = lane-16, K = 16..31 of the tile
// Gates use concatenated K: [W_hh (2048) | W_ih (70) | zero pad (26)].
// ---------------------------------------------------------------------------
__global__ void tlstm_conv_kernel(const float* __restrict__ Whh,
                                  const float* __restrict__ Wih,
                                  const float* __restrict__ Wc,
                                  unsigned short* __restrict__ tilesCat,
                                  unsigned short* __restrict__ tilesWc) {
  const int NT_CAT = RT_CAT * KT_CAT;   // 34304 tiles
  const int NT_WC  = RT_WC * KT_WC;     //  8192 tiles
  int gtid = blockIdx.x * blockDim.x + threadIdx.x;
  if (gtid >= (NT_CAT + NT_WC) * 32) return;
  int tile = gtid >> 5;
  int lane = gtid & 31;
  int nn   = lane & 15;
  int half = lane >> 4;

  __align__(16) unsigned short vals[16];
  unsigned short* dst;
  if (tile < NT_CAT) {
    int rt = tile / KT_CAT, kt = tile % KT_CAT;
    int row = rt * 16 + nn;
    int kb  = kt * 32 + half * 16;
    #pragma unroll
    for (int e = 0; e < 16; ++e) {
      int k = kb + e;
      float v = 0.0f;
      if (k < HIDDEN)                   v = Whh[(size_t)row * HIDDEN + k];
      else if (k < HIDDEN + INPUT_SIZE) v = Wih[(size_t)row * INPUT_SIZE + (k - HIDDEN)];
      vals[e] = f2bf(v);
    }
    dst = tilesCat + ((size_t)tile * 32 + lane) * 16;
  } else {
    int t2 = tile - NT_CAT;
    int rt = t2 / KT_WC, kt = t2 % KT_WC;
    int row = rt * 16 + nn;
    int kb  = kt * 32 + half * 16;
    #pragma unroll
    for (int e = 0; e < 16; ++e)
      vals[e] = f2bf(Wc[(size_t)row * HIDDEN + kb + e]);
    dst = tilesWc + ((size_t)t2 * 32 + lane) * 16;
  }
  *(uint4*)(dst)     = *(const uint4*)(vals);
  *(uint4*)(dst + 8) = *(const uint4*)(vals + 8);
}

// ---------------------------------------------------------------------------
// Kernel 2: init state: h0 = c0 = 0, stage x_0 into hx vector, zero pads,
// reset grid-barrier counter.
// ---------------------------------------------------------------------------
__global__ void tlstm_init_kernel(const float* __restrict__ X,
                                  float* __restrict__ c_f32,
                                  float* __restrict__ h_f32,
                                  unsigned short* __restrict__ hx,
                                  unsigned short* __restrict__ cvec,
                                  unsigned* __restrict__ cnt) {
  int tid = threadIdx.x;
  if (tid == 0) *cnt = 0u;
  for (int i = tid; i < HIDDEN; i += TPB) {
    c_f32[i] = 0.0f; h_f32[i] = 0.0f; hx[i] = 0; cvec[i] = 0;
  }
  for (int i = tid; i < KCAT - HIDDEN; i += TPB) {
    unsigned short v = 0;
    if (i < INPUT_SIZE) v = f2bf(X[(size_t)i * T_STEPS + 0]);  // X[i, t=0]
    hx[HIDDEN + i] = v;
  }
}

// ---------------------------------------------------------------------------
// Kernel 3: persistent scan kernel. 80 WGs x 256 threads = 640 waves.
// Waves 0..511  : one 16-row gate tile each (K = 2144 over hx vector)
// Waves 512..639: one 16-row Wc tile each   (K = 2048 over c vector)
// Per step: [WMMA GEMV] -> barrier -> [pointwise LSTM update + stage x_{t+1}]
//           -> barrier.
// ---------------------------------------------------------------------------
__global__ void __launch_bounds__(TPB, 1)
tlstm_persistent_kernel(const float* __restrict__ X,
                        const float* __restrict__ dt,
                        const float* __restrict__ b_ih,
                        const float* __restrict__ b_hh,
                        const float* __restrict__ bc,
                        const float* __restrict__ Wlin,
                        const float* __restrict__ blin,
                        const unsigned short* __restrict__ tilesCat,
                        const unsigned short* __restrict__ tilesWc,
                        float* __restrict__ gates,
                        float* __restrict__ cst,
                        float* __restrict__ c_f32,
                        float* __restrict__ h_f32,
                        unsigned short* __restrict__ hx,
                        unsigned short* __restrict__ cvec,
                        unsigned* __restrict__ cnt,
                        float* __restrict__ out) {
  __shared__ __align__(16) unsigned short svec[KCAT];
  __shared__ float red[TPB];

  const int tid    = threadIdx.x;
  const int wg     = blockIdx.x;
  const int lane   = tid & 31;
  const int gwave  = wg * WPW + (tid >> 5);
  const bool isWc  = (gwave >= RT_CAT);                // WGs 64..79 are all-Wc
  const int  rt    = isWc ? (gwave - RT_CAT) : gwave;
  const int  KT    = isWc ? KT_WC : KT_CAT;
  const unsigned short* tbase =
      isWc ? (tilesWc  + (size_t)rt * KT_WC  * TILE_ELEMS)
           : (tilesCat + (size_t)rt * KT_CAT * TILE_ELEMS);
  const unsigned short* gsrc = isWc ? cvec : hx;       // uniform per WG
  const int nChunk = ((isWc ? HIDDEN : KCAT) * 2) / 16; // uint4 chunks to LDS
  float* orow = isWc ? (cst + rt * 16) : (gates + rt * 16);
  // A-matrix lane layout (16-bit, 16x32): lanes<16 take K {0..15},
  // lanes>=16 take K {16..31} of each 32-wide chunk (matches B tiles).
  const int offA = (lane >> 4) * 16;
  const int gtid = wg * TPB + tid;

  unsigned phase = 0;
  for (int t = 0; t < T_STEPS; ++t) {
    // ---- stage vector (h||x or c) into LDS, coalesced 16B chunks ----
    {
      const uint4* s4 = (const uint4*)gsrc;
      uint4* d4 = (uint4*)svec;
      for (int i = tid; i < nChunk; i += TPB) d4[i] = s4[i];
    }
    __syncthreads();

    // ---- GEMV: 16 output rows per wave via v_wmma_f32_16x16x32_bf16 ----
    {
      const unsigned short* tp = tbase + (size_t)lane * 16;
      v8f acc0 = {};
      v8f acc1 = {};
      int kt = 0;
      for (; kt + 1 < KT; kt += 2) {
        v16bf b0 = *(const v16bf*)tp;
        union { v16bf v; v8bf h[2]; } A0;
        A0.h[0] = *(const v8bf*)(svec + kt * 32 + offA);
        A0.h[1] = *(const v8bf*)(svec + kt * 32 + offA + 8);
        acc0 = __builtin_amdgcn_wmma_f32_16x16x32_bf16(
                   false, A0.v, false, b0, (short)0, acc0, false, false);
        v16bf b1 = *(const v16bf*)(tp + TILE_ELEMS);
        union { v16bf v; v8bf h[2]; } A1;
        A1.h[0] = *(const v8bf*)(svec + (kt + 1) * 32 + offA);
        A1.h[1] = *(const v8bf*)(svec + (kt + 1) * 32 + offA + 8);
        acc1 = __builtin_amdgcn_wmma_f32_16x16x32_bf16(
                   false, A1.v, false, b1, (short)0, acc1, false, false);
        __builtin_prefetch(tp + 2 * TILE_ELEMS, 0, 1);   // global_prefetch
        tp += 2 * TILE_ELEMS;
      }
      for (; kt < KT; ++kt) {
        v16bf b0 = *(const v16bf*)tp;
        union { v16bf v; v8bf h[2]; } A0;
        A0.h[0] = *(const v8bf*)(svec + kt * 32 + offA);
        A0.h[1] = *(const v8bf*)(svec + kt * 32 + offA + 8);
        acc0 = __builtin_amdgcn_wmma_f32_16x16x32_bf16(
                   false, A0.v, false, b0, (short)0, acc0, false, false);
        tp += TILE_ELEMS;
      }
      // A rows are all identical (broadcast vector) -> D rows identical.
      // acc[0]: lanes 0..15 hold columns n = 0..15 of this row-tile.
      float v = acc0[0] + acc1[0];
      if (lane < 16) orow[lane] = v;
    }
    grid_barrier(cnt, ++phase);

    // ---- pointwise TLSTM update (threads 0..2047) + stage x_{t+1} ----
    if (gtid < HIDDEN) {
      const int j = gtid;
      float zi = gates[j]              + b_ih[j]              + b_hh[j];
      float zf = gates[j + HIDDEN]     + b_ih[j + HIDDEN]     + b_hh[j + HIDDEN];
      float zg = gates[j + 2 * HIDDEN] + b_ih[j + 2 * HIDDEN] + b_hh[j + 2 * HIDDEN];
      float zo = gates[j + 3 * HIDDEN] + b_ih[j + 3 * HIDDEN] + b_hh[j + 3 * HIDDEN];
      float si = sigmoidf_(zi);
      float sf = sigmoidf_(zf);
      float tg = tanhf(zg);
      float so = sigmoidf_(zo);
      float c_st = tanhf(cst[j] + bc[j]);
      float dec  = 1.0f / __logf(2.71828182845904523536f + dt[t] * (1.0f / 24.0f));
      float cold = c_f32[j];
      float cdec = cold - c_st + dec * c_st;
      float cnew = sf * cdec + si * tg;
      float hnew = so * tanhf(cnew);
      c_f32[j] = cnew;
      h_f32[j] = hnew;
      hx[j]    = f2bf(hnew);
      cvec[j]  = f2bf(cnew);
    } else if (gtid < HIDDEN + INPUT_SIZE && (t + 1) < T_STEPS) {
      const int i = gtid - HIDDEN;
      hx[HIDDEN + i] = f2bf(X[(size_t)i * T_STEPS + (t + 1)]);
    }
    grid_barrier(cnt, ++phase);
  }

  // ---- final: out = sigmoid(Wlin . h + blin), one WG reduction ----
  if (wg == 0) {
    float s = 0.0f;
    for (int k = tid; k < HIDDEN; k += TPB) s += Wlin[k] * h_f32[k];
    red[tid] = s;
    __syncthreads();
    for (int off = TPB / 2; off > 0; off >>= 1) {
      if (tid < off) red[tid] += red[tid + off];
      __syncthreads();
    }
    if (tid == 0) out[0] = sigmoidf_(red[0] + blin[0]);
  }
}

// ---------------------------------------------------------------------------
// Host launcher
// ---------------------------------------------------------------------------
extern "C" void kernel_launch(void* const* d_in, const int* in_sizes, int n_in,
                              void* d_out, int out_size, void* d_ws, size_t ws_size,
                              hipStream_t stream) {
  (void)in_sizes; (void)n_in; (void)out_size; (void)ws_size;

  const float* X    = (const float*)d_in[0];
  // d_in[1] Mask, d_in[2] Delta unused by the reference math
  const float* dt   = (const float*)d_in[3];
  const float* Wih  = (const float*)d_in[4];
  const float* bih  = (const float*)d_in[5];
  const float* Whh  = (const float*)d_in[6];
  const float* bhh  = (const float*)d_in[7];
  const float* Wc   = (const float*)d_in[8];
  const float* bc   = (const float*)d_in[9];
  const float* Wlin = (const float*)d_in[10];
  const float* blin = (const float*)d_in[11];
  float* out = (float*)d_out;

  // Workspace layout (256B aligned); total ~43.6 MB (fits the 192MB L2).
  char* ws = (char*)d_ws;
  size_t off = 0;
  auto take = [&](size_t bytes) -> char* {
    char* p = ws + off;
    off = (off + bytes + 255) & ~(size_t)255;
    return p;
  };
  unsigned*       cnt      = (unsigned*)take(sizeof(unsigned));
  float*          gates    = (float*)take((size_t)G4 * 4);
  float*          cst      = (float*)take((size_t)HIDDEN * 4);
  float*          c_f32    = (float*)take((size_t)HIDDEN * 4);
  float*          h_f32    = (float*)take((size_t)HIDDEN * 4);
  unsigned short* hx       = (unsigned short*)take((size_t)KCAT * 2);
  unsigned short* cvec     = (unsigned short*)take((size_t)HIDDEN * 2);
  unsigned short* tilesCat = (unsigned short*)take((size_t)RT_CAT * KT_CAT * TILE_ELEMS * 2);
  unsigned short* tilesWc  = (unsigned short*)take((size_t)RT_WC * KT_WC * TILE_ELEMS * 2);

  // 1) bf16 B-tile conversion (re-run every call: deterministic, amortized
  //    over 4096 steps; ~125MB of traffic, negligible at 23 TB/s)
  {
    const int totalThreads = (RT_CAT * KT_CAT + RT_WC * KT_WC) * 32;
    const int blocks = (totalThreads + TPB - 1) / TPB;
    tlstm_conv_kernel<<<blocks, TPB, 0, stream>>>(Whh, Wih, Wc, tilesCat, tilesWc);
  }
  // 2) state init (h0=c0=0, stage x_0, reset barrier counter)
  tlstm_init_kernel<<<1, TPB, 0, stream>>>(X, c_f32, h_f32, hx, cvec, cnt);
  // 3) persistent scan: 80 WGs co-resident, 2 grid barriers per timestep
  tlstm_persistent_kernel<<<NWG, TPB, 0, stream>>>(
      X, dt, bih, bhh, bc, Wlin, blin, tilesCat, tilesWc,
      gates, cst, c_f32, h_f32, hx, cvec, cnt, out);
}
